// ApplyKernel_22600117912289
// MI455X (gfx1250) — matrix operand order repaired and checked by reading, not compiled, not run on hardware
//
#include <hip/hip_runtime.h>

typedef __attribute__((ext_vector_type(16))) _Float16 v16h;
typedef __attribute__((ext_vector_type(8)))  float    v8f;
typedef __attribute__((ext_vector_type(2)))  float    v2f;

#define BZ   2
#define NPT  512
#define CIN  16
#define COUT 16
#define HID  64

// jax.nn.gelu(approximate=True):
// 0.5x(1+tanh(u)) with u = 0.79788456*(x+0.044715x^3)
// == x * (1 - 1/(exp(2u)+1))   (exact algebraic identity, correct limits)
__device__ __forceinline__ float gelu_fast(float x) {
    float u2 = x * (1.5957691216057308f + 0.07135481627260429f * x * x); // 2u
    float e  = __expf(u2);                          // hardware v_exp_f32
    return x * (1.0f - __builtin_amdgcn_rcpf(e + 1.0f));
}

__global__ __launch_bounds__(256)
void edgecond_wmma_kernel(const float* __restrict__ features,  // [2,512,16]
                          const float* __restrict__ geometry,  // [2,512,3]
                          const float* __restrict__ W1,        // [64,4]
                          const float* __restrict__ b1,        // [64]
                          const float* __restrict__ W2,        // [256,64]
                          const float* __restrict__ b2,        // [256]
                          float* __restrict__ out)             // [2,512,512,16]
{
    __shared__ float    sf[CIN];
    __shared__ float    sc[COUT];
    __shared__ float    sgb[4];
    __shared__ _Float16 sM[HID * COUT];   // M[h][i] f16

    const int b   = blockIdx.x;
    const int z   = blockIdx.y;
    const int tid = threadIdx.x;

    if (tid < CIN) sf[tid]  = features[(z * NPT + b) * CIN + tid];
    if (tid < 3)   sgb[tid] = geometry[(z * NPT + b) * 3 + tid];
    __syncthreads();

    // M[h][i] = sum_j W2[(i*16+j), h] * f[j]  (a-independent pre-contraction)
    for (int e = tid; e < HID * COUT; e += 256) {
        const int h = e >> 4;
        const int i = e & 15;
        float acc = 0.f;
        #pragma unroll
        for (int j = 0; j < CIN; ++j)
            acc += W2[(i * 16 + j) * HID + h] * sf[j];
        sM[h * COUT + i] = (_Float16)acc;
    }
    // c[i] = sum_j b2[i*16+j] * f[j]
    if (tid < COUT) {
        float acc = 0.f;
        #pragma unroll
        for (int j = 0; j < CIN; ++j)
            acc += b2[tid * 16 + j] * sf[j];
        sc[tid] = acc;
    }
    __syncthreads();

    const int lane = tid & 31;
    const int wave = tid >> 5;      // 0..7
    const int m    = lane & 15;     // row index (h_local for MLP1-A, pair for GEMM-A)
    const int half = lane >> 4;
    const int n    = m;             // output channel column

    // ---- MLP1 operands (f32 WMMA 16x16x4), block-invariant, kept in regs ----
    // A' (16h x 4): lane row = h_local, VGPR s -> K = 2*half+s
    // C  (16x16):   element r -> M = h = c*16 + 8*half + r, broadcast over N
    v2f Aw[4];
    v8f Cb[4];
    #pragma unroll
    for (int c = 0; c < 4; ++c) {
        Aw[c][0] = W1[(c * 16 + m) * 4 + 2 * half + 0];
        Aw[c][1] = W1[(c * 16 + m) * 4 + 2 * half + 1];
        #pragma unroll
        for (int r = 0; r < 8; ++r)
            Cb[c][r] = b1[c * 16 + 8 * half + r];
    }

    // ---- B operand of main f16 GEMM (32x16 per chunk), invariant per (z,b) ----
    // lane holds column n, element e -> K = half*8 + (e&7) + 16*(e>>3) (+32 for chunk 1)
    v16h B0, B1;
    #pragma unroll
    for (int e = 0; e < 16; ++e) {
        B0[e] = sM[(half * 8 + (e & 7) + 16 * (e >> 3))      * COUT + n];
        B1[e] = sM[(32 + half * 8 + (e & 7) + 16 * (e >> 3)) * COUT + n];
    }
    const float cb  = sc[n];
    const float gbx = sgb[0], gby = sgb[1], gbz = sgb[2];

    // Bias of the output contraction folded into the f16-WMMA C operand:
    // C[M,N] = c[n], constant over M and lane groups.
    v8f Dc;
    #pragma unroll
    for (int r = 0; r < 8; ++r) Dc[r] = cb;

    // 32-bit element offset base: out[z, 8*half, b, n]; a-row stride = NPT*COUT
    const unsigned obase = (((unsigned)z * NPT + 8u * half) * NPT + (unsigned)b) * COUT + n;

    for (int t = wave; t < NPT / 16; t += 8) {
        const int a = t * 16 + m;
        const float* gp = geometry + ((size_t)z * NPT + a) * 3;
        const float rx  = gbx - gp[0];
        const float ry  = gby - gp[1];
        const float rz  = gbz - gp[2];
        const float nrm = __builtin_amdgcn_sqrtf(rx * rx + ry * ry + rz * rz + 1e-12f);

        // B' = feat^T (4 x 16pairs): lane col = pair, VGPR s -> K = 2*half+s
        v2f Bf;
        Bf[0] = half ? rz : rx;
        Bf[1] = half ? nrm : ry;

        // H_chunk = W1_chunk x feat^T + b1_chunk   (fp32 matrix op, bias fused)
        v8f H0 = __builtin_amdgcn_wmma_f32_16x16x4_f32(false, Aw[0], false, Bf, (short)0, Cb[0], false, false);
        v8f H1 = __builtin_amdgcn_wmma_f32_16x16x4_f32(false, Aw[1], false, Bf, (short)0, Cb[1], false, false);
        v8f H2 = __builtin_amdgcn_wmma_f32_16x16x4_f32(false, Aw[2], false, Bf, (short)0, Cb[2], false, false);
        v8f H3 = __builtin_amdgcn_wmma_f32_16x16x4_f32(false, Aw[3], false, Bf, (short)0, Cb[3], false, false);

        // gelu + f16 pack: D' layout of f32 WMMA == A layout of f16 WMMA.
        // Process each H chunk contiguously so packed f32 ops stay in-place.
        v16h A0, A1;
        #pragma unroll
        for (int r = 0; r < 8; ++r) A0[r]     = (_Float16)gelu_fast(H0[r]);
        #pragma unroll
        for (int r = 0; r < 8; ++r) A0[8 + r] = (_Float16)gelu_fast(H1[r]);
        #pragma unroll
        for (int r = 0; r < 8; ++r) A1[r]     = (_Float16)gelu_fast(H2[r]);
        #pragma unroll
        for (int r = 0; r < 8; ++r) A1[8 + r] = (_Float16)gelu_fast(H3[r]);

        // out_tile(16a x 16i) = H(16a x 64h) x M(64h x 16i) + c
        v8f D = __builtin_amdgcn_wmma_f32_16x16x32_f16(false, A0, false, B0, (short)0, Dc, false, false);
        D     = __builtin_amdgcn_wmma_f32_16x16x32_f16(false, A1, false, B1, (short)0, D,  false, false);

        // element r -> out row a = t*16 + r + 8*half; byte offset r*32768 fits IOFFSET
        float* po = out + (obase + (unsigned)t * (16u * NPT * COUT));
        #pragma unroll
        for (int r = 0; r < 8; ++r)
            po[r * (NPT * COUT)] = D[r];
    }
}

extern "C" void kernel_launch(void* const* d_in, const int* in_sizes, int n_in,
                              void* d_out, int out_size, void* d_ws, size_t ws_size,
                              hipStream_t stream) {
    const float* features = (const float*)d_in[0];
    const float* geometry = (const float*)d_in[1];
    const float* W1       = (const float*)d_in[2];
    const float* b1       = (const float*)d_in[3];
    const float* W2       = (const float*)d_in[4];
    const float* b2       = (const float*)d_in[5];
    float* out = (float*)d_out;

    dim3 grid(NPT, BZ);
    dim3 block(256);
    edgecond_wmma_kernel<<<grid, block, 0, stream>>>(features, geometry,
                                                     W1, b1, W2, b2, out);
}
